// MHA_31155692765475
// MI455X (gfx1250) — matrix operand order repaired and checked by reading, not compiled, not run on hardware
//
#include <hip/hip_runtime.h>

typedef __bf16 bf16;
typedef __attribute__((ext_vector_type(16))) __bf16 v16bf;
typedef __attribute__((ext_vector_type(8)))  float  v8f;

#define HIDDEN   1024
#define HEADS    16
#define HEAD_DIM 64
#define BATCH    4
#define SEQ      2048
#define NROW     (BATCH * SEQ)   // 8192

// ---------------------------------------------------------------------------
// Fragment-ready global layout: each WMMA fragment = 32 lanes x 16 bf16,
// stored contiguously (1024B). A lane reads its 32B slice with 2x b128.
// ---------------------------------------------------------------------------
__device__ inline v16bf load_frag_g(const bf16* frag_base) {
  const int lane = threadIdx.x & 31;
  return *(const v16bf*)(frag_base + lane * 16);
}

// A-fragment (16x32 bf16) from LDS, row-major with stride ld (elements).
// Per lane the 16 elements form two contiguous 8-element (16B) runs.
__device__ inline v16bf load_a_frag_lds(const bf16* base, int ld) {
  const int lane = threadIdx.x & 31;
  const int m    = lane & 15;
  const int half = lane >> 4;
  v16bf a;
#pragma unroll
  for (int e = 0; e < 8; ++e) {
    a[e]     = base[m * ld + half * 16 + e];           // k = half*16 + e
    a[e + 8] = base[m * ld + half * 16 + 8 + e];       // k = half*16 + 8 + e
  }
  return a;
}

__device__ inline v8f wmma_bf16(v16bf a, v16bf b, v8f c) {
  return __builtin_amdgcn_wmma_f32_16x16x32_bf16(
      false, a, false, b, (short)0, c, false, false);
}

__device__ inline void wait_lds() {
  asm volatile("s_wait_dscnt 0x0" ::: "memory");
}

// ---------------------------------------------------------------------------
// Swizzle index helpers (element index within each swizzled buffer)
// ---------------------------------------------------------------------------
// Generic A-frag layout over [8192 rows][1024 cols]: [mtile 512][ks 32][lane][16]
__device__ inline size_t a_swz_off(int gm, int gn) {
  const int mtile = gm >> 4, m = gm & 15;
  const int ks = gn >> 5, kk = gn & 31;
  const int half = (kk >> 3) & 1;
  const int e = (kk & 7) | ((kk >> 4) << 3);
  const int lane = half * 16 + m;
  return (((size_t)mtile * 32 + ks) * 32 + lane) * 16 + e;
}
// Q per (b,h): A-frag tiles [b*16+h][ltile 128][ks 2][lane][16]
__device__ inline size_t q_swz_off(int gm, int gn) {
  const int b = gm >> 11, l = gm & (SEQ - 1);
  const int h = gn >> 6, d = gn & 63;
  const int ltile = l >> 4, m = l & 15;
  const int ks = d >> 5, kk = d & 31;
  const int half = (kk >> 3) & 1;
  const int e = (kk & 7) | ((kk >> 4) << 3);
  const int lane = half * 16 + m;
  return ((((size_t)(b * HEADS + h) * 128 + ltile) * 2 + ks) * 32 + lane) * 16 + e;
}
// K per (b,h): B-frag tiles of K^T (k=dim, n=seq): [bh][ltile 128][ks 2][lane][16]
__device__ inline size_t k_swz_off(int gm, int gn) {
  const int b = gm >> 11, l = gm & (SEQ - 1);
  const int h = gn >> 6, d = gn & 63;
  const int ltile = l >> 4, n = l & 15;
  const int ks = d >> 5, dd = d & 31;
  const int lane = ((dd >> 3) & 1) * 16 + n;
  const int e = (dd & 7) | ((dd >> 4) << 3);
  return ((((size_t)(b * HEADS + h) * 128 + ltile) * 2 + ks) * 32 + lane) * 16 + e;
}
// V per (b,h): B-frag tiles (k=seq, n=dim): [bh][lblk 64][ni 4][lane][16]
__device__ inline size_t v_swz_off(int gm, int gn) {
  const int b = gm >> 11, l = gm & (SEQ - 1);
  const int h = gn >> 6, d = gn & 63;
  const int lblk = l >> 5, lr = l & 31;
  const int ni = d >> 4, n = d & 15;
  const int lane = (lr >> 4) * 16 + n;
  const int e = lr & 15;
  return ((((size_t)(b * HEADS + h) * 64 + lblk) * 4 + ni) * 32 + lane) * 16 + e;
}

// ---------------------------------------------------------------------------
// Convert + swizzle kernels (one-time, bandwidth trivial)
// ---------------------------------------------------------------------------
__global__ __launch_bounds__(256) void cvt_swz_x_kernel(const float* __restrict__ x,
                                                        bf16* __restrict__ out) {
  const int idx = blockIdx.x * 256 + threadIdx.x;      // 1,048,576 runs of 8
  const int rhalf = idx & 1;
  const int lane  = (idx >> 1) & 31;
  const int ks    = (idx >> 6) & 31;
  const int mtile = idx >> 11;
  const int m     = mtile * 16 + (lane & 15);
  const int half  = lane >> 4;
  const int kbase = ks * 32 + rhalf * 16 + half * 8;
  const float* src = x + (size_t)m * HIDDEN + kbase;
  bf16* dst = out + (size_t)(idx >> 1) * 16 + rhalf * 8;
#pragma unroll
  for (int j = 0; j < 8; ++j) dst[j] = (bf16)src[j];
}

__global__ __launch_bounds__(256) void cvt_swz_w_kernel(const float* __restrict__ w,
                                                        bf16* __restrict__ out) {
  const int idx = blockIdx.x * 256 + threadIdx.x;      // 1,048,576 elements
  const int e     = idx & 15;
  const int lane  = (idx >> 4) & 31;
  const int ks    = (idx >> 9) & 31;
  const int ntile = idx >> 14;
  const int n = ntile * 16 + (lane & 15);
  const int k = ks * 32 + (e >> 3) * 16 + (lane >> 4) * 8 + (e & 7);
  out[idx] = (bf16)w[(size_t)k * HIDDEN + n];
}

// ---------------------------------------------------------------------------
// GEMM: C[8192,1024] = A @ W + bias.  A,W in frag-ready layouts, no LDS.
// MODE: 0 = f32 out, 1 = Q-swz bf16, 2 = K-swz bf16, 3 = V-swz bf16.
// ---------------------------------------------------------------------------
template <int MODE>
__global__ __launch_bounds__(256) void gemm_swz_kernel(
    const bf16* __restrict__ A, const bf16* __restrict__ W,
    const float* __restrict__ bias, void* __restrict__ out) {
  const int lane = threadIdx.x & 31;
  const int wave = threadIdx.x >> 5;
  const int wm = wave >> 1;   // 0..3
  const int wn = wave & 1;    // 0..1

  const v8f vzero = {0.f, 0.f, 0.f, 0.f, 0.f, 0.f, 0.f, 0.f};
  v8f c[2][4];
#pragma unroll
  for (int mi = 0; mi < 2; ++mi)
#pragma unroll
    for (int ni = 0; ni < 4; ++ni) c[mi][ni] = vzero;

  const int mtile0 = blockIdx.x * 8 + wm * 2;
  const int ntile0 = blockIdx.y * 8 + wn * 4;

  for (int ks = 0; ks < 32; ++ks) {
    v16bf a0 = load_frag_g(A + (((size_t)mtile0 * 32 + ks) * 32) * 16);
    v16bf a1 = load_frag_g(A + (((size_t)(mtile0 + 1) * 32 + ks) * 32) * 16);
#pragma unroll
    for (int ni = 0; ni < 4; ++ni) {
      v16bf b = load_frag_g(W + (((size_t)(ntile0 + ni) * 32 + ks) * 32) * 16);
      c[0][ni] = wmma_bf16(a0, b, c[0][ni]);
      c[1][ni] = wmma_bf16(a1, b, c[1][ni]);
    }
  }

  const int half = lane >> 4;
  const int nl   = lane & 15;
#pragma unroll
  for (int mi = 0; mi < 2; ++mi)
#pragma unroll
    for (int ni = 0; ni < 4; ++ni) {
      const int gn = blockIdx.y * 128 + wn * 64 + ni * 16 + nl;
      const float bv = bias[gn];
#pragma unroll
      for (int r = 0; r < 8; ++r) {
        const int gm = blockIdx.x * 128 + wm * 32 + mi * 16 + r + 8 * half;
        const float val = c[mi][ni][r] + bv;
        if (MODE == 0)      ((float*)out)[(size_t)gm * HIDDEN + gn] = val;
        else if (MODE == 1) ((bf16*)out)[q_swz_off(gm, gn)] = (bf16)val;
        else if (MODE == 2) ((bf16*)out)[k_swz_off(gm, gn)] = (bf16)val;
        else                ((bf16*)out)[v_swz_off(gm, gn)] = (bf16)val;
      }
    }
}

// ---------------------------------------------------------------------------
// Flash attention. Grid = 512 (64 (b,h) x 8 query chunks), 8 waves/block.
// Wave owns 32 query rows == 32 lanes -> softmax done row-per-lane through
// an LDS transpose bounce; no cross-lane butterflies.
// ---------------------------------------------------------------------------
__global__ __launch_bounds__(256) void attn_kernel(
    const bf16* __restrict__ Q, const bf16* __restrict__ K,
    const bf16* __restrict__ V, bf16* __restrict__ O) {
  __shared__ float Sf[8][32][36];  // 144B row stride: 16B-aligned, <=2-way conflicts
  __shared__ bf16  Ps[8][32][40];  // 80B row stride: 16B-aligned runs for A-frags

  const int lane = threadIdx.x & 31;
  const int wave = threadIdx.x >> 5;
  const int half = lane >> 4;
  const int nl   = lane & 15;

  const int bh = blockIdx.x >> 3;
  const int b  = bh >> 4;
  const int h  = bh & 15;
  const int q0 = (blockIdx.x & 7) * 256 + wave * 32;

  const bf16* qh = Q + (size_t)bh * (128 * 2 * 32 * 16);
  const bf16* kh = K + (size_t)bh * (128 * 2 * 32 * 16);
  const bf16* vh = V + (size_t)bh * (64 * 4 * 32 * 16);

  v16bf aq[2][2];
#pragma unroll
  for (int mi = 0; mi < 2; ++mi)
#pragma unroll
    for (int ks = 0; ks < 2; ++ks)
      aq[mi][ks] = load_frag_g(qh + (((size_t)((q0 >> 4) + mi) * 2 + ks) * 32) * 16);

  const v8f vzero = {0.f, 0.f, 0.f, 0.f, 0.f, 0.f, 0.f, 0.f};
  v8f o[2][4];
#pragma unroll
  for (int mi = 0; mi < 2; ++mi)
#pragma unroll
    for (int ni = 0; ni < 4; ++ni) o[mi][ni] = vzero;

  float mx = -3.0e38f;   // running max for row (q0 + lane)
  float sm = 0.0f;       // running sum for row (q0 + lane)
  const float scale = 0.125f;  // 1/sqrt(64)

  for (int kb = 0; kb < SEQ; kb += 32) {
    // ---- S = Q K^T (32x32) ----
    v8f s[2][2];
    s[0][0] = vzero; s[0][1] = vzero; s[1][0] = vzero; s[1][1] = vzero;
#pragma unroll
    for (int ks = 0; ks < 2; ++ks) {
      v16bf bk0 = load_frag_g(kh + (((size_t)((kb >> 4) + 0) * 2 + ks) * 32) * 16);
      v16bf bk1 = load_frag_g(kh + (((size_t)((kb >> 4) + 1) * 2 + ks) * 32) * 16);
      s[0][0] = wmma_bf16(aq[0][ks], bk0, s[0][0]);
      s[0][1] = wmma_bf16(aq[0][ks], bk1, s[0][1]);
      s[1][0] = wmma_bf16(aq[1][ks], bk0, s[1][0]);
      s[1][1] = wmma_bf16(aq[1][ks], bk1, s[1][1]);
    }

    // ---- transpose bounce: C-layout scatter -> row-major rows ----
#pragma unroll
    for (int mi = 0; mi < 2; ++mi)
#pragma unroll
      for (int ni = 0; ni < 2; ++ni)
#pragma unroll
        for (int r = 0; r < 8; ++r)
          Sf[wave][mi * 16 + r + 8 * half][ni * 16 + nl] = s[mi][ni][r];
    wait_lds();

    // ---- row-per-lane softmax: lane owns query row (q0 + lane) ----
    float ev[32];
    {
      const float4* srow = (const float4*)(&Sf[wave][lane][0]);
#pragma unroll
      for (int c2 = 0; c2 < 8; ++c2) {
        const float4 v4 = srow[c2];
        ev[4 * c2 + 0] = v4.x * scale;
        ev[4 * c2 + 1] = v4.y * scale;
        ev[4 * c2 + 2] = v4.z * scale;
        ev[4 * c2 + 3] = v4.w * scale;
      }
    }
    float rmax = ev[0];
#pragma unroll
    for (int j = 1; j < 32; ++j) rmax = fmaxf(rmax, ev[j]);
    const float mnew = fmaxf(mx, rmax);
    const float corr = __expf(mx - mnew);
    mx = mnew;

    float rsum = 0.0f;
    unsigned int pk[16];
#pragma unroll
    for (int j = 0; j < 16; ++j) {
      const float p0 = __expf(ev[2 * j]     - mnew);
      const float p1 = __expf(ev[2 * j + 1] - mnew);
      rsum += p0 + p1;
      const unsigned int u0 = __builtin_bit_cast(unsigned short, (bf16)p0);
      const unsigned int u1 = __builtin_bit_cast(unsigned short, (bf16)p1);
      pk[j] = u0 | (u1 << 16);
    }
    sm = sm * corr + rsum;

    {  // packed row-major P store (A-frag-friendly)
      uint4* prow = (uint4*)(&Ps[wave][lane][0]);
#pragma unroll
      for (int c2 = 0; c2 < 4; ++c2) {
        uint4 u;
        u.x = pk[4 * c2 + 0]; u.y = pk[4 * c2 + 1];
        u.z = pk[4 * c2 + 2]; u.w = pk[4 * c2 + 3];
        prow[c2] = u;
      }
    }

    // ---- broadcast corr to C-layout rows, rescale O ----
#pragma unroll
    for (int mi = 0; mi < 2; ++mi)
#pragma unroll
      for (int r = 0; r < 8; ++r) {
        const float cr = __shfl(corr, mi * 16 + r + 8 * half, 32);
#pragma unroll
        for (int ni = 0; ni < 4; ++ni) o[mi][ni][r] *= cr;
      }
    wait_lds();

    // ---- O += P @ V  (32x32 @ 32x64) ----
    v16bf ap0 = load_a_frag_lds(&Ps[wave][0][0], 40);
    v16bf ap1 = load_a_frag_lds(&Ps[wave][16][0], 40);
#pragma unroll
    for (int ni = 0; ni < 4; ++ni) {
      v16bf bv = load_frag_g(vh + (((size_t)(kb >> 5) * 4 + ni) * 32) * 16);
      o[0][ni] = wmma_bf16(ap0, bv, o[0][ni]);
      o[1][ni] = wmma_bf16(ap1, bv, o[1][ni]);
    }
  }

  // ---- normalize + write attention output (A-frag-ready for O-projection) ----
  const float inv = 1.0f / sm;
#pragma unroll
  for (int mi = 0; mi < 2; ++mi)
#pragma unroll
    for (int r = 0; r < 8; ++r) {
      const float iv = __shfl(inv, mi * 16 + r + 8 * half, 32);
      const int gm = b * SEQ + q0 + mi * 16 + r + 8 * half;
#pragma unroll
      for (int ni = 0; ni < 4; ++ni) {
        const int gn = h * HEAD_DIM + ni * 16 + nl;
        O[a_swz_off(gm, gn)] = (bf16)(o[mi][ni][r] * iv);
      }
    }
}

// ---------------------------------------------------------------------------
// Launch
// ---------------------------------------------------------------------------
extern "C" void kernel_launch(void* const* d_in, const int* in_sizes, int n_in,
                              void* d_out, int out_size, void* d_ws, size_t ws_size,
                              hipStream_t stream) {
  const float* x  = (const float*)d_in[0];
  const float* wq = (const float*)d_in[1];
  const float* bq = (const float*)d_in[2];
  const float* wk = (const float*)d_in[3];
  const float* bk = (const float*)d_in[4];
  const float* wv = (const float*)d_in[5];
  const float* bv = (const float*)d_in[6];
  const float* wo = (const float*)d_in[7];
  const float* bo = (const float*)d_in[8];
  float* out = (float*)d_out;

  const size_t nX = (size_t)NROW * HIDDEN;     // 8,388,608
  const size_t nW = (size_t)HIDDEN * HIDDEN;   // 1,048,576

  char* p = (char*)d_ws;
  bf16* xs  = (bf16*)p; p += nX * sizeof(bf16);   // x, A-frag-ready
  bf16* wqs = (bf16*)p; p += nW * sizeof(bf16);   // weights, B-frag-ready
  bf16* wks = (bf16*)p; p += nW * sizeof(bf16);
  bf16* wvs = (bf16*)p; p += nW * sizeof(bf16);
  bf16* wos = (bf16*)p; p += nW * sizeof(bf16);
  bf16* qs  = (bf16*)p; p += nX * sizeof(bf16);   // Q, per-head A-frag-ready
  bf16* ks_ = (bf16*)p; p += nX * sizeof(bf16);   // K, per-head B^T-frag-ready
  bf16* vs  = (bf16*)p; p += nX * sizeof(bf16);   // V, per-head B-frag-ready
  bf16* os  = (bf16*)p; p += nX * sizeof(bf16);   // attn out, A-frag-ready

  cvt_swz_x_kernel<<<4096, 256, 0, stream>>>(x, xs);
  cvt_swz_w_kernel<<<4096, 256, 0, stream>>>(wq, wqs);
  cvt_swz_w_kernel<<<4096, 256, 0, stream>>>(wk, wks);
  cvt_swz_w_kernel<<<4096, 256, 0, stream>>>(wv, wvs);
  cvt_swz_w_kernel<<<4096, 256, 0, stream>>>(wo, wos);

  dim3 gGemm(NROW / 128, HIDDEN / 128);
  gemm_swz_kernel<1><<<gGemm, 256, 0, stream>>>(xs, wqs, bq, qs);
  gemm_swz_kernel<2><<<gGemm, 256, 0, stream>>>(xs, wks, bk, ks_);
  gemm_swz_kernel<3><<<gGemm, 256, 0, stream>>>(xs, wvs, bv, vs);

  attn_kernel<<<512, 256, 0, stream>>>(qs, ks_, vs, os);

  gemm_swz_kernel<0><<<gGemm, 256, 0, stream>>>(os, wos, bo, out);
}
